// BehaviorSpecificPFF_89936615178802
// MI455X (gfx1250) — matrix operand order repaired and checked by reading, not compiled, not run on hardware
//
#include <hip/hip_runtime.h>

// ---------------------------------------------------------------------------
// BehaviorSpecificPFF for MI455X (gfx1250, wave32, WMMA + async LDS copies)
//
// Dense multi-behavior FFN; per-token select at the end; behavior 0 (padding)
// handled by pre-zeroing d_out.
//
// Compute: bf16 WMMA 16x16x32, f32 accumulation.
// Data movement: W1/W2 chunks double-buffered in LDS via
// GLOBAL_LOAD_ASYNC_TO_LDS_B128 (ASYNCcnt), overlapped with WMMA compute.
// ---------------------------------------------------------------------------

#define TOKENS   16384        // B*T = 8*2048
#define DM       512          // d_model
#define DF       2048         // d_ff
#define NB       4
#define LN_EPS   1e-5f

#define M_TILE   128          // tokens per block
#define FCHUNK   32           // d_ff chunk per iteration (double-buffered)
#define NCHUNK   (DF / FCHUNK)
#define THREADS  512          // 16 wave32s

typedef __attribute__((ext_vector_type(16))) __bf16 v16bf;
typedef __attribute__((ext_vector_type(8)))  __bf16 v8bf;
typedef __attribute__((ext_vector_type(8)))  float  v8f;

// Dynamic LDS pool layout (bytes)
#define XOFF   0              // bf16 [128][512] x tile            = 131072
#define W1OFF  131072         // 2 x bf16 [32][512] W1 chunks      =  65536
#define W2OFF  196608         // 2 x bf16 [512][32] W2 chunks      =  65536
#define HOFF   262144         // bf16 [128][32] hidden chunk       =   8192
#define LDS_BYTES 270336
#define WBUF_BYTES 32768      // one W1 or W2 chunk buffer
// Epilogue reuses bytes [0, 262144) as f32 r[128][512].

// ---- CDNA5 async global->LDS copy (ASYNCcnt) -------------------------------
// GVS mode: mem = SADDR(64b) + VADDR(32b); VDST holds the LDS byte address.
__device__ __forceinline__ void async_b128(unsigned lds_byte, unsigned voff_byte,
                                           const void* sbase) {
  asm volatile("global_load_async_to_lds_b128 %0, %1, %2"
               :: "v"(lds_byte), "v"(voff_byte), "s"(sbase)
               : "memory");
}
__device__ __forceinline__ void wait_async0() {
  asm volatile("s_wait_asynccnt 0" ::: "memory");
}

// ---- WMMA fragment loaders (CDNA5 16-bit layouts, wave32) -----------------
// A (16x32 bf16): lane m = lane&15; K = kb..kb+7 and kb+16..kb+23,
// kb = (lane>>4)*8 -> two 16B LDS loads.
__device__ __forceinline__ v16bf ldsA(const __bf16* base, int stride,
                                      int mtile, int ks, int lane) {
  const __bf16* p = base + (size_t)(mtile * 16 + (lane & 15)) * stride
                         + ks * 32 + ((lane >> 4) << 3);
  v8bf lo = *(const v8bf*)p;
  v8bf hi = *(const v8bf*)(p + 16);
  return __builtin_shufflevector(lo, hi,
      0, 1, 2, 3, 4, 5, 6, 7, 8, 9, 10, 11, 12, 13, 14, 15);
}

// B (32x16 bf16): lane n = lane&15; K = kb..kb+7 and kb+16..kb+23,
// kb = (lane>>4)*8 -> two 16B LDS loads.
__device__ __forceinline__ v16bf ldsB(const __bf16* base, int stride,
                                      int ntile, int ks, int lane) {
  const __bf16* p = base + (size_t)(ntile * 16 + (lane & 15)) * stride
                         + ks * 32 + ((lane >> 4) << 3);
  v8bf lo = *(const v8bf*)p;
  v8bf hi = *(const v8bf*)(p + 16);
  return __builtin_shufflevector(lo, hi,
      0, 1, 2, 3, 4, 5, 6, 7, 8, 9, 10, 11, 12, 13, 14, 15);
}

// ---- prologue kernels ------------------------------------------------------
__global__ void convert_f32_bf16(const float* __restrict__ s,
                                 __bf16* __restrict__ d, int n4) {
  int i = blockIdx.x * blockDim.x + threadIdx.x;
  if (i < n4) {
    float4 v = ((const float4*)s)[i];
    int o = i * 4;
    d[o + 0] = (__bf16)v.x; d[o + 1] = (__bf16)v.y;
    d[o + 2] = (__bf16)v.z; d[o + 3] = (__bf16)v.w;
  }
}

__global__ void zero_out_kernel(float* __restrict__ o, int n4) {
  int i = blockIdx.x * blockDim.x + threadIdx.x;
  if (i < n4) ((float4*)o)[i] = make_float4(0.f, 0.f, 0.f, 0.f);
}

// ---- fused PFF kernel ------------------------------------------------------
__global__ __launch_bounds__(THREADS, 1)
void pff_fused(const float* __restrict__ xg, const int* __restrict__ bseq,
               const __bf16* __restrict__ w1b, const float* __restrict__ b1g,
               const __bf16* __restrict__ w2b, const float* __restrict__ b2g,
               const float* __restrict__ gammag, const float* __restrict__ betag,
               float* __restrict__ outg) {
  extern __shared__ char smem[];
  __bf16* xs  = (__bf16*)(smem + XOFF);
  __bf16* w1s = (__bf16*)(smem + W1OFF);
  __bf16* w2s = (__bf16*)(smem + W2OFF);
  __bf16* hs  = (__bf16*)(smem + HOFF);
  float*  rs  = (float*)smem;       // epilogue reuse

  const int tid  = threadIdx.x;
  const int lane = tid & 31;
  const int wid  = tid >> 5;        // 0..15
  const int n    = blockIdx.y;      // behavior 0..3 <-> b_seq value n+1
  const int t0   = blockIdx.x * M_TILE;

  // Early-exit if no token in this tile selects behavior n+1.
  __shared__ int s_any;
  if (tid == 0) s_any = 0;
  __syncthreads();
  if (tid < M_TILE && bseq[t0 + tid] == n + 1) s_any = 1;
  __syncthreads();
  if (!s_any) return;

  const __bf16* w1src = w1b + (size_t)n * DF * DM;
  const __bf16* w2src = w2b + (size_t)n * DM * DF;
  const float*  b1p   = b1g + (size_t)n * DF;

  // LDS byte addresses (low 32 bits of generic LDS pointers).
  const unsigned w1l0 = (unsigned)(size_t)(void*)w1s;
  const unsigned w2l0 = (unsigned)(size_t)(void*)w2s;

  // Issue async copies for W1/W2 chunk `fcn` into buffer fcn&1.
  // Each wave copies 4x 512B slices of each 32KB buffer (8 async B128 ops).
  auto issue_chunk = [&](int fcn) {
    const int bufn = fcn & 1;
    const int f0n  = fcn * FCHUNK;
    const unsigned g0 = (unsigned)(wid * 128 + lane);      // 16B granule id
    const char* g1 = (const char*)(w1src + (size_t)f0n * DM);  // contiguous
    const char* g2 = (const char*)(w2src + f0n);               // row stride DF
    #pragma unroll
    for (int i = 0; i < 4; ++i) {
      unsigned g = g0 + i * 32;
      async_b128(w1l0 + bufn * WBUF_BYTES + g * 16, g * 16, g1);
    }
    #pragma unroll
    for (int i = 0; i < 4; ++i) {
      unsigned g = g0 + i * 32;                  // row = g>>2, col16 = g&3
      async_b128(w2l0 + bufn * WBUF_BYTES + g * 16,
                 (g >> 2) * (DF * 2) + (g & 3) * 16, g2);
    }
  };

  // Prefetch chunk 0, then stage x (fp32->bf16) while it flies.
  issue_chunk(0);
  {
    const float4* sx = (const float4*)(xg + (size_t)t0 * DM);
    #pragma unroll 4
    for (int g = tid; g < (M_TILE * DM) / 4; g += THREADS) {
      float4 v = sx[g];
      int o = g * 4;
      xs[o + 0] = (__bf16)v.x; xs[o + 1] = (__bf16)v.y;
      xs[o + 2] = (__bf16)v.z; xs[o + 3] = (__bf16)v.w;
    }
  }

  const int mrow  = wid & 7;        // m-tile (8 of them: 128 tokens)
  const int ftile = wid >> 3;       // GEMM1: f-tile within chunk (0/1)
  const int dhalf = wid >> 3;       // GEMM2: d columns [0,256) or [256,512)

  const v8f vzero = {0.f, 0.f, 0.f, 0.f, 0.f, 0.f, 0.f, 0.f};
  v8f acc[16];
  #pragma unroll
  for (int i = 0; i < 16; ++i) acc[i] = vzero;

  #pragma unroll 1
  for (int fc = 0; fc < NCHUNK; ++fc) {
    const int buf = fc & 1;
    wait_async0();        // this wave's copies for chunk fc have landed
    __syncthreads();      // all waves' copies landed; prior reads of buf done

    if (fc + 1 < NCHUNK) issue_chunk(fc + 1);   // overlap with compute below

    // GEMM1: h[128x32] = relu(x[128x512] @ W1chunk^T + b1).
    // Each wave: 1 of 16 (8m x 2f) tiles, K=512 -> 16 wmma.
    {
      const __bf16* w1c = w1s + buf * (FCHUNK * DM);
      v8f h0 = vzero;
      #pragma unroll
      for (int ks = 0; ks < DM / 32; ++ks) {
        v16bf a = ldsA(xs, DM, mrow, ks, lane);
        v16bf b = ldsB(w1c, DM, ftile, ks, lane);
        h0 = __builtin_amdgcn_wmma_f32_16x16x32_bf16(false, a, false, b,
                                                     (short)0, h0, false, false);
      }
      // D layout: lane n = lane&15, VGPR r -> M = r + 8*(lane>>4).
      int fcl = ftile * 16 + (lane & 15);
      float bias = b1p[fc * FCHUNK + fcl];
      #pragma unroll
      for (int r = 0; r < 8; ++r) {
        int row = mrow * 16 + r + ((lane >> 4) << 3);
        hs[row * FCHUNK + fcl] = (__bf16)fmaxf(h0[r] + bias, 0.0f);
      }
    }
    __syncthreads();

    // GEMM2: y[128x512] += h[128x32] @ W2chunk^T.
    // Each wave: 1 m-tile x 16 d-tiles, K=32 -> 16 wmma, A-frag reused 16x.
    {
      const __bf16* w2c = w2s + buf * (DM * FCHUNK);
      v16bf a = ldsA(hs, FCHUNK, mrow, 0, lane);
      #pragma unroll
      for (int dt = 0; dt < 16; ++dt) {
        v16bf b = ldsB(w2c, FCHUNK, dhalf * 16 + dt, 0, lane);
        acc[dt] = __builtin_amdgcn_wmma_f32_16x16x32_bf16(false, a, false, b,
                                                (short)0, acc[dt], false, false);
      }
    }
  }

  __syncthreads();
  // Spill y accumulators to LDS as f32 (reuse staging region).
  #pragma unroll
  for (int dt = 0; dt < 16; ++dt) {
    int dcol = dhalf * 256 + dt * 16 + (lane & 15);
    #pragma unroll
    for (int r = 0; r < 8; ++r) {
      int row = mrow * 16 + r + ((lane >> 4) << 3);
      rs[row * DM + dcol] = acc[dt][r];
    }
  }
  __syncthreads();

  // Epilogue: residual + b2, LayerNorm (wave32 shfl reduction), gamma/beta,
  // per-token select. One wave handles 8 token rows.
  const float* b2p = b2g    + (size_t)n * DM;
  const float* gp  = gammag + (size_t)n * DM;
  const float* bp  = betag  + (size_t)n * DM;
  #pragma unroll 1
  for (int j8 = 0; j8 < 8; ++j8) {
    int row = wid * 8 + j8;
    int t = t0 + row;
    float vals[16];
    float sum = 0.f, ss = 0.f;
    #pragma unroll
    for (int j = 0; j < 16; ++j) {
      int d = lane + j * 32;
      float v = rs[row * DM + d] + xg[(size_t)t * DM + d] + b2p[d];
      vals[j] = v; sum += v; ss += v * v;
    }
    #pragma unroll
    for (int off = 16; off > 0; off >>= 1) {
      sum += __shfl_xor(sum, off, 32);
      ss  += __shfl_xor(ss,  off, 32);
    }
    float mean = sum * (1.0f / DM);
    float var  = ss * (1.0f / DM) - mean * mean;
    float rstd = rsqrtf(var + LN_EPS);
    if (bseq[t] == n + 1) {
      #pragma unroll
      for (int j = 0; j < 16; ++j) {
        int d = lane + j * 32;
        outg[(size_t)t * DM + d] = (vals[j] - mean) * rstd * gp[d] + bp[d];
      }
    }
  }
}

// ---- launcher --------------------------------------------------------------
extern "C" void kernel_launch(void* const* d_in, const int* in_sizes, int n_in,
                              void* d_out, int out_size, void* d_ws, size_t ws_size,
                              hipStream_t stream) {
  (void)in_sizes; (void)n_in; (void)ws_size;
  const float* x     = (const float*)d_in[0];
  const int*   bseq  = (const int*)d_in[1];
  const float* W1    = (const float*)d_in[2];
  const float* b1    = (const float*)d_in[3];
  const float* W2    = (const float*)d_in[4];
  const float* b2    = (const float*)d_in[5];
  const float* gamma = (const float*)d_in[6];
  const float* beta  = (const float*)d_in[7];
  float* out = (float*)d_out;

  // Workspace: bf16 copies of W1 and W2 (8 MB each).
  __bf16* w1b = (__bf16*)d_ws;
  __bf16* w2b = (__bf16*)((char*)d_ws + (size_t)NB * DF * DM * sizeof(__bf16));

  const int nW4 = (NB * DF * DM) / 4;
  convert_f32_bf16<<<(nW4 + 255) / 256, 256, 0, stream>>>(W1, w1b, nW4);
  convert_f32_bf16<<<(nW4 + 255) / 256, 256, 0, stream>>>(W2, w2b, nW4);

  const int z4 = out_size / 4;
  zero_out_kernel<<<(z4 + 255) / 256, 256, 0, stream>>>(out, z4);

  (void)hipFuncSetAttribute((const void*)pff_fused,
                            hipFuncAttributeMaxDynamicSharedMemorySize,
                            LDS_BYTES);

  dim3 grid(TOKENS / M_TILE, NB);
  pff_fused<<<grid, THREADS, LDS_BYTES, stream>>>(x, bseq, w1b, b1, w2b, b2,
                                                  gamma, beta, out);
}
